// StandardTransformer_75127567941685
// MI455X (gfx1250) — matrix operand order repaired and checked by reading, not compiled, not run on hardware
//
#include <hip/hip_runtime.h>
#include <cstdint>
#include <cstddef>

using bhalf = __bf16;
typedef bhalf v16bf __attribute__((ext_vector_type(16)));
typedef float v8f  __attribute__((ext_vector_type(8)));

#define WMMA_BF16(a, b, c) \
  __builtin_amdgcn_wmma_f32_16x16x32_bf16(false, (a), false, (b), (short)0, (c), false, false)

__device__ inline v8f zero8() {
  v8f z = {0.f, 0.f, 0.f, 0.f, 0.f, 0.f, 0.f, 0.f};
  return z;
}

// ---------------- embedding + sinusoidal posenc (f32 + bf16 mirror) ----------
__global__ void k_embed(const int* __restrict__ ids, const float* __restrict__ emb,
                        float* __restrict__ x, bhalf* __restrict__ xb) {
  int t = blockIdx.x;      // token index 0..B*S-1
  int d = threadIdx.x;     // 0..255
  int s = t & 2047;        // S = 2048
  int id = ids[t];
  int i2 = d & ~1;
  float w = __expf(-(float)i2 * (9.210340371976184f / 256.0f));  // ln(10000)/D
  float ang = (float)s * w;
  float pe = (d & 1) ? __cosf(ang) : __sinf(ang);
  float v = emb[id * 256 + d] * 16.0f + pe;                      // sqrt(256)=16
  x[(size_t)t * 256 + d]  = v;
  xb[(size_t)t * 256 + d] = (bhalf)v;
}

// ---------------- additive key-padding mask: 0 or -1e30 ----------------
__global__ void k_mask(const int* __restrict__ ids, float* __restrict__ maskadd) {
  int i = blockIdx.x * 256 + threadIdx.x;
  maskadd[i] = (ids[i] != 0) ? 0.0f : -1e30f;
}

// ---------------- f32 [K,N] -> bf16 transposed [N,K] ----------------
__global__ void k_conv_t(const float* __restrict__ W, bhalf* __restrict__ Wt,
                         int K, int N) {
  int i = blockIdx.x * 256 + threadIdx.x;
  if (i >= K * N) return;
  int k = i / N, n = i - k * N;
  Wt[(size_t)n * K + k] = (bhalf)W[i];
}

// ---------------- GEMM: C[M,N] = op(A[M,K] @ W[K,N] + bias) ----------------
// A is bf16 row-major [M,K]; Wt is bf16 pre-transposed [N,K].
// Block = 128 threads = 4 waves; each wave computes a 16x64 tile (A-fragment
// reused across 4 WMMAs). grid = (M/16, N/256).
// OM (output mode): 0 = f32 row-major, 1 = bf16 row-major,
//                   2 = bf16 per-head transposed V layout [B,H,64,S]
//                       (assumes S=2048, d_k=64, N=256)
template <bool RELU, int OM>
__global__ void k_gemm(const bhalf* __restrict__ A, const bhalf* __restrict__ Wt,
                       const float* __restrict__ bias, void* __restrict__ Cout,
                       int M, int N, int K) {
  const int lane  = threadIdx.x & 31;
  const int wave  = threadIdx.x >> 5;
  const int mtile = blockIdx.x;
  const int ncol0 = (blockIdx.y * 4 + wave) * 64;
  const int l15   = lane & 15;
  const int kb8   = (lane & 16) ? 8 : 0;    // unused: A-fragment also uses kb16
  const int kb16  = (lane & 16) ? 16 : 0;   // B-fragment K base
  const int row   = mtile * 16 + l15;
  const bhalf* arow = A + (size_t)row * K;

  v8f acc[4];
#pragma unroll
  for (int n = 0; n < 4; ++n) acc[n] = zero8();

  for (int k0 = 0; k0 < K; k0 += 32) {
    v16bf af;
#pragma unroll
    for (int i = 0; i < 8; ++i) {
      af[i]     = arow[k0 + kb16 + i];
      af[i + 8] = arow[k0 + kb16 + 8 + i];
    }
    __builtin_prefetch(arow + k0 + 256, 0, 0);  // global_prefetch_b8
#pragma unroll
    for (int n = 0; n < 4; ++n) {
      const bhalf* wp = Wt + (size_t)(ncol0 + n * 16 + l15) * K + k0 + kb16;
      v16bf bfrag;
#pragma unroll
      for (int i = 0; i < 16; ++i) bfrag[i] = wp[i];
      acc[n] = WMMA_BF16(af, bfrag, acc[n]);
    }
  }
  const int prow = mtile * 16 + ((lane & 16) ? 8 : 0);  // C layout: M=r (+8 hi half)
#pragma unroll
  for (int n = 0; n < 4; ++n) {
    int col = ncol0 + n * 16 + l15;
    float bv = bias[col];
#pragma unroll
    for (int r = 0; r < 8; ++r) {
      float v = acc[n][r] + bv;
      if (RELU) v = fmaxf(v, 0.0f);
      int rr = prow + r;
      if (OM == 0) {
        ((float*)Cout)[(size_t)rr * N + col] = v;
      } else if (OM == 1) {
        ((bhalf*)Cout)[(size_t)rr * N + col] = (bhalf)v;
      } else {
        // V^T layout: token rr = (b,s), col = (h,d) -> [((b*4+h)*64+d)*2048 + s]
        int bb = rr >> 11, ss = rr & 2047;
        int hh = col >> 6, dd = col & 63;
        ((bhalf*)Cout)[(((size_t)bb * 4 + hh) * 64 + dd) * 2048 + ss] = (bhalf)v;
      }
    }
  }
}

// ---------------- fused flash attention (transposed-score formulation) -------
// One wave per (b, h, 16-query tile).  S^T = K @ Q^T so the C-layout puts the
// query in the lane and the key in the VGPR index:
//   * softmax stats = in-lane reduce over 16 regs + ONE shfl_xor(16) each
//   * running m/l and 1/l are per-lane scalars
//   * O^T = V^T @ P^T keeps lane=query, so corr rescale is a per-lane scalar
//   * P^T B-fragment built with 8 half-wave exchanges (no LDS transpose)
//   * entire softmax runs in the exp2 domain (scale premultiplied by log2e),
//     so every exponential is a bare v_exp_f32 with no log2e multiply
// Masked keys use an additive -1e30: after any real key has been seen they
// underflow to 0 in exp2; before that, their bogus contributions are wiped
// because the first real key forces corr = exp2(-huge) = 0 on l and acc.
__global__ void k_attention(const bhalf* __restrict__ Q, const bhalf* __restrict__ Km,
                            const bhalf* __restrict__ Vt, const float* __restrict__ maskadd,
                            bhalf* __restrict__ Out) {
  const int lane = threadIdx.x & 31;
  const int wave = threadIdx.x >> 5;
  const int w    = blockIdx.x * 8 + wave;
  const int qt   = w & 127;          // S/16 = 128 query tiles
  const int h    = (w >> 7) & 3;     // 4 heads
  const int b    = w >> 9;           // batch
  const int S = 2048, D = 256;
  const float SC2 = 0.125f * 1.4426950408889634f;  // (1/sqrt(d_k)) * log2(e)

  const int  l15   = lane & 15;
  const int  kb8   = (lane & 16) ? 8 : 0;
  const int  kb16  = (lane & 16) ? 16 : 0;
  const int  prow8 = (lane & 16) ? 8 : 0;
  const bool hihalf = (lane & 16) != 0;

  // Q^T B-fragments (fixed over the key loop): lane = query col, elems run d
  const bhalf* qp = Q + (size_t)(b * S + qt * 16 + l15) * D + h * 64;
  v16bf qB[2];
#pragma unroll
  for (int kk = 0; kk < 2; ++kk)
#pragma unroll
    for (int i = 0; i < 16; ++i) qB[kk][i] = qp[kk * 32 + kb8 + (i & 7) + (i >> 3) * 16];

  float m = -1e30f, l = 0.0f;
  v8f acc[4];
#pragma unroll
  for (int n = 0; n < 4; ++n) acc[n] = zero8();

  const float* mrow = maskadd + b * S;
  const bhalf* vbase = Vt + ((size_t)b * 4 + h) * 64 * S;

  for (int kb = 0; kb < S / 32; ++kb) {
    const int key0 = kb * 32;
    // ---- S^T for 32 keys: two 16-key C fragments, K-dim = d (2 x 32) ----
    v8f s0 = zero8(), s1 = zero8();
#pragma unroll
    for (int kk = 0; kk < 2; ++kk) {
      const bhalf* kp0 = Km + (size_t)(b * S + key0 + l15) * D + h * 64 + kk * 32 + kb8;
      const bhalf* kp1 = kp0 + (size_t)16 * D;
      v16bf a0, a1;
#pragma unroll
      for (int i = 0; i < 8; ++i) {
        a0[i] = kp0[i]; a0[i + 8] = kp0[16 + i];
        a1[i] = kp1[i]; a1[i + 8] = kp1[16 + i];
      }
      s0 = WMMA_BF16(a0, qB[kk], s0);
      s1 = WMMA_BF16(a1, qB[kk], s1);
    }
    // ---- exp2-domain scale + additive pad mask (key = key0 + prow8 + r) ----
    const float* mp0 = mrow + key0 + prow8;
#pragma unroll
    for (int r = 0; r < 8; ++r) {
      s0[r] = s0[r] * SC2 + mp0[r];
      s1[r] = s1[r] * SC2 + mp0[16 + r];
    }
    // ---- per-query (per-lane) online softmax ----
    float mx = s0[0];
#pragma unroll
    for (int r = 1; r < 8; ++r) mx = fmaxf(mx, s0[r]);
#pragma unroll
    for (int r = 0; r < 8; ++r) mx = fmaxf(mx, s1[r]);
    mx = fmaxf(mx, __shfl_xor(mx, 16));       // other 16 keys live in partner half
    float nm   = fmaxf(m, mx);
    float corr = __builtin_amdgcn_exp2f(m - nm);
    float p0[8], p1[8];
    float rs = 0.0f;
#pragma unroll
    for (int r = 0; r < 8; ++r) {
      p0[r] = __builtin_amdgcn_exp2f(s0[r] - nm);
      p1[r] = __builtin_amdgcn_exp2f(s1[r] - nm);
      rs += p0[r] + p1[r];
    }
    rs += __shfl_xor(rs, 16);
    l = l * corr + rs;
    m = nm;
#pragma unroll
    for (int n = 0; n < 4; ++n)
#pragma unroll
      for (int r = 0; r < 8; ++r) acc[n][r] *= corr;

    // ---- build P^T B-fragment: one half-wave exchange per register ----
    // lanes<16 own keys 0..7 (p0) & 16..23 (p1); partner holds 8..15 & 24..31
    v16bf pb;
#pragma unroll
    for (int r = 0; r < 8; ++r) {
      float send = hihalf ? p0[r] : p1[r];
      float xt   = __shfl_xor(send, 16);
      float lo   = hihalf ? xt    : p0[r];    // elems 0..7: keys base+0..7
      float hv   = hihalf ? p1[r] : xt;       // elems 8..15: keys base+8..15
      pb[r]     = (bhalf)lo;
      pb[r + 8] = (bhalf)hv;
    }
    // ---- O^T += V^T @ P^T  (A = V^T fragment from [B,H,64,S], contiguous) ----
#pragma unroll
    for (int n = 0; n < 4; ++n) {
      const bhalf* vp = vbase + (size_t)(n * 16 + l15) * S + key0 + kb16;
      v16bf va;
#pragma unroll
      for (int i = 0; i < 8; ++i) {
        va[i]     = vp[i];
        va[i + 8] = vp[8 + i];
      }
      acc[n] = WMMA_BF16(va, pb, acc[n]);
    }
  }
  // ---- epilogue: O^T C-layout -> Out[token, d] (bf16); lane = query ----
  float inv = 1.0f / fmaxf(l, 1e-30f);
  bhalf* op = Out + (size_t)(b * S + qt * 16 + l15) * D + h * 64 + prow8;
#pragma unroll
  for (int n = 0; n < 4; ++n)
#pragma unroll
    for (int r = 0; r < 8; ++r)
      op[n * 16 + r] = (bhalf)(acc[n][r] * inv);   // 8x bf16 contiguous runs
}

// ---- fused residual add + LayerNorm (one wave per row); writes f32 + bf16 ----
__global__ void k_add_ln(float* __restrict__ x, const float* __restrict__ a,
                         const float* __restrict__ g, const float* __restrict__ be,
                         bhalf* __restrict__ xb) {
  const int lane  = threadIdx.x & 31;
  const int wave  = threadIdx.x >> 5;
  const size_t row = (size_t)blockIdx.x * 8 + wave;
  float* xr = x + row * 256;
  bhalf* xbr = xb + row * 256;
  const float* ar = a + row * 256;
  float v[8];
  float s = 0.f;
#pragma unroll
  for (int i = 0; i < 8; ++i) { v[i] = xr[lane * 8 + i] + ar[lane * 8 + i]; s += v[i]; }
  s += __shfl_xor(s, 1); s += __shfl_xor(s, 2); s += __shfl_xor(s, 4);
  s += __shfl_xor(s, 8); s += __shfl_xor(s, 16);
  float mu = s * (1.0f / 256.0f);
  float t = 0.f;
#pragma unroll
  for (int i = 0; i < 8; ++i) { float d = v[i] - mu; t += d * d; }
  t += __shfl_xor(t, 1); t += __shfl_xor(t, 2); t += __shfl_xor(t, 4);
  t += __shfl_xor(t, 8); t += __shfl_xor(t, 16);
  float rstd = rsqrtf(t * (1.0f / 256.0f) + 1e-5f);
#pragma unroll
  for (int i = 0; i < 8; ++i) {
    float o = (v[i] - mu) * rstd * g[lane * 8 + i] + be[lane * 8 + i];
    xr[lane * 8 + i]  = o;
    xbr[lane * 8 + i] = (bhalf)o;
  }
}

// ---------------- masked mean-pool + classifier ----------------
__global__ void k_pool_cls(const float* __restrict__ x, const int* __restrict__ ids,
                           const float* __restrict__ Wc, const float* __restrict__ bc,
                           float* __restrict__ out) {
  __shared__ float pooled[256];
  const int b = blockIdx.x, d = threadIdx.x;
  const float* xb = x + (size_t)b * 2048 * 256;
  const int* idb = ids + b * 2048;
  float s = 0.f, cnt = 0.f;
  for (int t = 0; t < 2048; ++t) {
    if (idb[t] != 0) { s += xb[(size_t)t * 256 + d]; cnt += 1.f; }
  }
  pooled[d] = s / fmaxf(cnt, 1e-9f);
  __syncthreads();
  if (d < 10) {
    float acc = bc[d];
    for (int k = 0; k < 256; ++k) acc += pooled[k] * Wc[k * 10 + d];
    out[b * 10 + d] = acc;
  }
}

// ---------------- host launcher ----------------
extern "C" void kernel_launch(void* const* d_in, const int* in_sizes, int n_in,
                              void* d_out, int out_size, void* d_ws, size_t ws_size,
                              hipStream_t stream) {
  (void)in_sizes; (void)n_in; (void)out_size; (void)ws_size;
  const int B = 16, S = 2048, D = 256, F = 1024, L = 4;
  const size_t M = (size_t)B * S;

  const int*   ids = (const int*)d_in[0];
  const float* emb = (const float*)d_in[1];
  const float* Wq  = (const float*)d_in[2];
  const float* bq  = (const float*)d_in[3];
  const float* Wk  = (const float*)d_in[4];
  const float* bk  = (const float*)d_in[5];
  const float* Wv  = (const float*)d_in[6];
  const float* bv  = (const float*)d_in[7];
  const float* Wo  = (const float*)d_in[8];
  const float* bo  = (const float*)d_in[9];
  const float* g1  = (const float*)d_in[10];
  const float* be1 = (const float*)d_in[11];
  const float* W1  = (const float*)d_in[12];
  const float* b1  = (const float*)d_in[13];
  const float* W2  = (const float*)d_in[14];
  const float* b2  = (const float*)d_in[15];
  const float* g2  = (const float*)d_in[16];
  const float* be2 = (const float*)d_in[17];
  const float* Wc  = (const float*)d_in[18];
  const float* bc  = (const float*)d_in[19];

  // workspace carving (~225 MB total)
  char* p = (char*)d_ws;
  auto carve = [&](size_t bytes) -> void* {
    void* r = (void*)p;
    p += (bytes + 255) & ~(size_t)255;
    return r;
  };
  float* x    = (float*)carve(M * D * sizeof(float));
  float* t1   = (float*)carve(M * D * sizeof(float));
  float* madd = (float*)carve(M * sizeof(float));
  bhalf* xb   = (bhalf*)carve(M * D * sizeof(bhalf));  // bf16 mirror of x
  bhalf* t0b  = (bhalf*)carve(M * D * sizeof(bhalf));  // attention output (bf16)
  bhalf* qb   = (bhalf*)carve(M * D * sizeof(bhalf));
  bhalf* kbuf = (bhalf*)carve(M * D * sizeof(bhalf));
  bhalf* vtb  = (bhalf*)carve(M * D * sizeof(bhalf));  // V^T per head [B,H,64,S]
  bhalf* hb   = (bhalf*)carve(M * F * sizeof(bhalf));
  bhalf* wqt  = (bhalf*)carve((size_t)L * D * D * sizeof(bhalf));
  bhalf* wkt  = (bhalf*)carve((size_t)L * D * D * sizeof(bhalf));
  bhalf* wvt  = (bhalf*)carve((size_t)L * D * D * sizeof(bhalf));
  bhalf* wot  = (bhalf*)carve((size_t)L * D * D * sizeof(bhalf));
  bhalf* w1t  = (bhalf*)carve((size_t)L * D * F * sizeof(bhalf));
  bhalf* w2t  = (bhalf*)carve((size_t)L * D * F * sizeof(bhalf));

  k_embed<<<B * S, 256, 0, stream>>>(ids, emb, x, xb);
  k_mask<<<(int)(M / 256), 256, 0, stream>>>(ids, madd);

  const int gDD = (D * D + 255) / 256, gDF = (D * F + 255) / 256;
  for (int l0 = 0; l0 < L; ++l0) {
    size_t oDD = (size_t)l0 * D * D, oDF = (size_t)l0 * D * F;
    k_conv_t<<<gDD, 256, 0, stream>>>(Wq + oDD, wqt + oDD, D, D);
    k_conv_t<<<gDD, 256, 0, stream>>>(Wk + oDD, wkt + oDD, D, D);
    k_conv_t<<<gDD, 256, 0, stream>>>(Wv + oDD, wvt + oDD, D, D);
    k_conv_t<<<gDD, 256, 0, stream>>>(Wo + oDD, wot + oDD, D, D);
    k_conv_t<<<gDF, 256, 0, stream>>>(W1 + oDF, w1t + oDF, D, F);
    k_conv_t<<<gDF, 256, 0, stream>>>(W2 + oDF, w2t + oDF, F, D);
  }

  dim3 blk(128);
  dim3 gD((int)(M / 16), D / 256);  // N=256 -> grid.y=1
  dim3 gF((int)(M / 16), F / 256);  // N=1024 -> grid.y=4
  const int attnBlocks = (B * 4 * (S / 16)) / 8;

  for (int l0 = 0; l0 < L; ++l0) {
    size_t oDD = (size_t)l0 * D * D, oDF = (size_t)l0 * D * F;
    size_t oD = (size_t)l0 * D, oF = (size_t)l0 * F;
    k_gemm<false, 1><<<gD, blk, 0, stream>>>(xb,  wqt + oDD, bq + oD, qb,   (int)M, D, D);
    k_gemm<false, 1><<<gD, blk, 0, stream>>>(xb,  wkt + oDD, bk + oD, kbuf, (int)M, D, D);
    k_gemm<false, 2><<<gD, blk, 0, stream>>>(xb,  wvt + oDD, bv + oD, vtb,  (int)M, D, D);
    k_attention<<<attnBlocks, 256, 0, stream>>>(qb, kbuf, vtb, madd, t0b);
    k_gemm<false, 0><<<gD, blk, 0, stream>>>(t0b, wot + oDD, bo + oD, t1,   (int)M, D, D);
    k_add_ln<<<(int)(M / 8), 256, 0, stream>>>(x, t1, g1 + oD, be1 + oD, xb);
    k_gemm<true,  1><<<gF, blk, 0, stream>>>(xb,  w1t + oDF, b1 + oF, hb,   (int)M, F, D);
    k_gemm<false, 0><<<gD, blk, 0, stream>>>(hb,  w2t + oDF, b2 + oD, t1,   (int)M, D, F);
    k_add_ln<<<(int)(M / 8), 256, 0, stream>>>(x, t1, g2 + oD, be2 + oD, xb);
  }

  k_pool_cls<<<B, 256, 0, stream>>>(x, ids, Wc, bc, (float*)d_out);
}
